// RNNModel_46231027974476
// MI455X (gfx1250) — matrix operand order repaired and checked by reading, not compiled, not run on hardware
//
#include <hip/hip_runtime.h>

// ---------------------------------------------------------------------------
// LSTM world-model forward for MI455X (gfx1250), wave32 + WMMA bf16.
//   phase 1: build bf16 inputs/weights, x_gates GEMM (HBM-bound, ~25us)
//   phase 2: persistent LSTM kernel, 32 WGs, W_hh staged in LDS via TDM,
//            grid split-barrier per timestep, xg prefetch for t+1
//   phase 3: output GEMM + fused bias + log_softmax epilogue (HBM-bound)
// ---------------------------------------------------------------------------

typedef __bf16 bf16_t;
typedef __attribute__((ext_vector_type(16))) __bf16 v16bf;
typedef __attribute__((ext_vector_type(8)))  float  v8f;
typedef __attribute__((ext_vector_type(4)))  unsigned v4u;
typedef __attribute__((ext_vector_type(8)))  int      v8i;
typedef __attribute__((ext_vector_type(4)))  int      v4i;

#define B_SZ     64
#define T_SZ     1024
#define ZD       64
#define KIN      96          // 68 input features padded to 3 k-tiles of 32
#define RNN_D    512
#define NG       2048        // 4*RNN
#define NOUT     961
#define NOUT_P   976         // padded to 61 n-tiles of 16
#define MROWS    65472       // 64 * 1023
#define NWG_LSTM 32

#define LSTM_LDS_WHH   0                     // 64*512 bf16 = 65536 B
#define LSTM_LDS_GATES 65536                 // 64*64  f32  = 16384 B
#define LSTM_LDS_CST   (65536 + 16384)       // 64*16  f32  =  4096 B
#define LSTM_LDS_BYTES (65536 + 16384 + 4096)

#if defined(__has_builtin)
#if __has_builtin(__builtin_amdgcn_tensor_load_to_lds) && \
    __has_builtin(__builtin_amdgcn_s_wait_tensorcnt)
#define HAVE_TDM 1
#endif
#endif
#ifndef HAVE_TDM
#define HAVE_TDM 0
#endif

// ---------------- WMMA helpers (wave32, contiguous k: lane half h holds k = 16h..16h+15) -------

static __device__ __forceinline__ v8f wmma_bf16(v16bf a, v16bf b, v8f c) {
  return __builtin_amdgcn_wmma_f32_16x16x32_bf16(false, a, false, b,
                                                 (short)0, c, false, false);
}

// A fragment: 16x32 bf16 tile, row-major source with stride lda.
// lane<16: M=lane, K=0..15; lane>=16: M=lane-16, K=16..31.
static __device__ __forceinline__ v16bf load_a(const bf16_t* A, int lda) {
  const int lane = threadIdx.x & 31;
  const int half = lane >> 4;
  const int m    = lane & 15;
  const bf16_t* row = A + (size_t)m * lda + half * 16;
  v16bf f;
#pragma unroll
  for (int v = 0; v < 8; ++v) {
    const int kb = 2 * v;
    f[2 * v]     = row[kb];
    f[2 * v + 1] = row[kb + 1];
  }
  return f;
}

// B fragment: 32x16 bf16 tile gathered as W^T from row-major W[n][k].
// lane<16: K=0..15 ; lane>=16: K=16..31 ; N = lane&15.  16 contiguous loads.
static __device__ __forceinline__ v16bf load_bT(const bf16_t* W, int ldw) {
  const int lane = threadIdx.x & 31;
  const int n    = lane & 15;
  const int kh   = (lane >> 4) * 16;
  const bf16_t* p = W + (size_t)n * ldw + kh;
  v16bf f;
#pragma unroll
  for (int e = 0; e < 16; ++e) f[e] = p[e];
  return f;
}

// ---------------- device-wide split barrier (persistent LSTM) --------------

static __device__ __forceinline__ void gridsync(unsigned* sync, unsigned nwg) {
  __syncthreads();
  if (threadIdx.x == 0) {
    __builtin_amdgcn_fence(__ATOMIC_RELEASE, "agent");  // global_wb path
    unsigned gen = __hip_atomic_load(&sync[1], __ATOMIC_RELAXED,
                                     __HIP_MEMORY_SCOPE_AGENT);
    unsigned a = __hip_atomic_fetch_add(&sync[0], 1u, __ATOMIC_RELAXED,
                                        __HIP_MEMORY_SCOPE_AGENT);
    if (a == nwg - 1) {
      __hip_atomic_store(&sync[0], 0u, __ATOMIC_RELAXED,
                         __HIP_MEMORY_SCOPE_AGENT);
      __hip_atomic_store(&sync[1], gen + 1u, __ATOMIC_RELEASE,
                         __HIP_MEMORY_SCOPE_AGENT);
    } else {
      while (__hip_atomic_load(&sync[1], __ATOMIC_RELAXED,
                               __HIP_MEMORY_SCOPE_AGENT) == gen)
        __builtin_amdgcn_s_sleep(8);
    }
    __builtin_amdgcn_fence(__ATOMIC_ACQUIRE, "agent");  // global_inv path
  }
  __syncthreads();
}

// ---------------- prep kernels ---------------------------------------------

__global__ void prep_inp(const float* z, const int* actions, const float* embed,
                         bf16_t* inp) {
  const size_t row = blockIdx.x;
  const int c = threadIdx.x;
  float v;
  if (c < ZD)            v = z[row * ZD + c];
  else if (c < ZD + 4)   v = embed[(size_t)actions[row] * 4 + (c - ZD)];
  else                   v = 0.f;
  inp[row * KIN + c] = (bf16_t)v;
}

__global__ void cvt_pad(const float* src, bf16_t* dst, int srows, int scols,
                        int dcols) {
  const int r = blockIdx.x;
  const int c = threadIdx.x;
  float v = (r < srows && c < scols) ? src[(size_t)r * scols + c] : 0.f;
  dst[(size_t)r * dcols + c] = (bf16_t)v;
}

__global__ void prep_bias(const float* bih, const float* bhh, float* bcomb) {
  const int i = blockIdx.x * 256 + threadIdx.x;
  if (i < NG) bcomb[i] = bih[i] + bhh[i];
}

__global__ void init_state(bf16_t* hbuf, unsigned* sync) {
  const int i = blockIdx.x * 256 + threadIdx.x;
  if (i < 2 * B_SZ * RNN_D) hbuf[i] = (bf16_t)0.f;
  if (i < 2) sync[i] = 0u;
}

// ---------------- phase 1: x_gates = inp @ W_ih^T + (b_ih + b_hh) ----------
// grid (1024, 16), block 256 (8 waves).  WG tile 64(M) x 128(N), K=96.
__global__ void xgates_gemm(const bf16_t* inp, const bf16_t* wih,
                            const float* bcomb, float* xg) {
  const int wave = threadIdx.x >> 5;
  const int lane = threadIdx.x & 31;
  const int half = lane >> 4;
  const int n    = lane & 15;
  const size_t m0 = (size_t)blockIdx.x * 64;
  const int n0 = blockIdx.y * 128 + wave * 16;   // one n-tile per wave

  v8f acc[4] = {v8f{}, v8f{}, v8f{}, v8f{}};
  for (int k = 0; k < KIN; k += 32) {
    v16bf bf = load_bT(wih + (size_t)n0 * KIN + k, KIN);
#pragma unroll
    for (int mt = 0; mt < 4; ++mt) {
      v16bf af = load_a(inp + (m0 + mt * 16) * KIN + k, KIN);
      acc[mt] = wmma_bf16(af, bf, acc[mt]);
    }
  }
  const int col = n0 + n;
  const float bias = bcomb[col];
#pragma unroll
  for (int mt = 0; mt < 4; ++mt)
#pragma unroll
    for (int r = 0; r < 8; ++r)
      xg[(m0 + mt * 16 + r + half * 8) * NG + col] = acc[mt][r] + bias;
}

// ---------------- phase 2: persistent LSTM ---------------------------------
// 32 WGs x 256 threads. WG owns hidden units [j0, j0+16): its 64 gate rows
// {g*512 + j0 + jj} are TDM-staged into LDS once (loop-invariant; LDS is not
// invalidated by the per-step agent fences). Per step: [64x512]@[512x64]
// via 256 WMMAs (B from LDS), cell update, h_next published, grid barrier.
__global__ void lstm_kernel(const bf16_t* whh, const float* xg,
                            const int* dones, bf16_t* hbuf, bf16_t* hs,
                            unsigned* sync) {
  extern __shared__ __align__(16) unsigned char smem[];
  bf16_t* whh_s = (bf16_t*)(smem + LSTM_LDS_WHH);   // [64][512], row=g*16+jj
  float*  gates = (float*)(smem + LSTM_LDS_GATES);  // [batch][g*16+jj]
  float*  cst   = (float*)(smem + LSTM_LDS_CST);    // carry c, f32

  const int j0   = blockIdx.x * 16;
  const int tid  = threadIdx.x;
  const int wave = tid >> 5;
  const int lane = tid & 31;
  const int half = lane >> 4;
  const int n    = lane & 15;
  const int mt   = wave >> 1;        // batch tile 0..3
  const int ntb  = (wave & 1) * 2;   // gate tiles {ntb, ntb+1}

#if HAVE_TDM
  // --- Tensor Data Mover: stage the 4x16x512 bf16 gate-slice into LDS ------
  // x: 512 elems/row, y: 16 hidden units (stride 512 rows' worth? no: stride
  // 512 elems = 1 row), z: 4 gates (stride 512*512 elems = 512 rows).
  if (tid < 32) {            // one wave issues; EXEC ignored by TDM
    const unsigned long long ga =
        (unsigned long long)(const void*)whh + (unsigned long long)j0 * RNN_D * 2;
    const unsigned lds_addr = (unsigned)(unsigned long long)(void*)whh_s;
    v4u g0;
    g0[0] = 1u;                                       // count=1, user D#
    g0[1] = lds_addr;                                 // [63:32] lds_addr
    g0[2] = (unsigned)(ga & 0xFFFFFFFFull);           // [95:64] global lo
    g0[3] = (unsigned)((ga >> 32) & 0x1FFFFFFull)     // [120:96] global hi
          | (2u << 30);                               // [127:126] type=2
    v8i g1;
    g1[0] = (int)(1u << 16);          // wg_mask=0, data_size=1 (2 bytes)
    g1[1] = (int)(512u << 16);        // tensor_dim0 = 512 (lo16 @ [63:48])
    g1[2] = (int)(2048u << 16);       // dim0 hi=0; tensor_dim1=2048 (lo16)
    g1[3] = (int)(512u << 16);        // dim1 hi=0; tile_dim0 = 512
    g1[4] = (int)((4u << 16) | 16u);  // tile_dim1=16, tile_dim2=4
    g1[5] = (int)512;                 // tensor_dim0_stride lo32 = 512 elems
    g1[6] = 0;                        // stride0 hi16=0; stride1 lo16=0
    g1[7] = (int)(0x40000u >> 16);    // tensor_dim1_stride = 512*512 elems
    v4i g2;
    g2[0] = 4;                        // tensor_dim2 = 4
    g2[1] = 0; g2[2] = 0; g2[3] = 0;
    v4i g3; g3[0] = 0; g3[1] = 0; g3[2] = 0; g3[3] = 0;
#if __clang_major__ >= 23
    v8i g4 = {0, 0, 0, 0, 0, 0, 0, 0};
    __builtin_amdgcn_tensor_load_to_lds(g0, g1, g2, g3, g4, 0);
#else
    __builtin_amdgcn_tensor_load_to_lds(g0, g1, g2, g3, 0);
#endif
    __builtin_amdgcn_s_wait_tensorcnt(0);
  }
#else
  for (int i = tid; i < 64 * RNN_D; i += 256) {
    const int r = i >> 9, kk = i & 511;
    const int g = r >> 4, jj = r & 15;
    whh_s[i] = whh[((size_t)(g * RNN_D + j0 + jj)) * RNN_D + kk];
  }
#endif
  for (int i = tid; i < 64 * 16; i += 256) cst[i] = 0.f;
  __syncthreads();

  for (int t = 0; t < T_SZ; ++t) {
    const bf16_t* hcur = hbuf + (size_t)(t & 1) * (B_SZ * RNN_D);
    bf16_t*       hnxt = hbuf + (size_t)((t + 1) & 1) * (B_SZ * RNN_D);

    v8f c0 = {}; v8f c1 = {};
    const int r0 = (ntb * 16 + n) * RNN_D;        // LDS rows for B tiles
    const int r1 = ((ntb + 1) * 16 + n) * RNN_D;
    for (int k = 0; k < RNN_D; k += 32) {
      v16bf af = load_a(hcur + (size_t)(mt * 16) * RNN_D + k, RNN_D);
      const int kb = k + half * 16;
      v16bf b0, b1;
#pragma unroll
      for (int e = 0; e < 16; ++e) {              // ds_load_b128 x2 per tile
        b0[e] = whh_s[r0 + kb + e];
        b1[e] = whh_s[r1 + kb + e];
      }
      c0 = wmma_bf16(af, b0, c0);
      c1 = wmma_bf16(af, b1, c1);
    }
#pragma unroll
    for (int r = 0; r < 8; ++r) {
      gates[(mt * 16 + r + half * 8) * 64 + ntb * 16 + n]       = c0[r];
      gates[(mt * 16 + r + half * 8) * 64 + (ntb + 1) * 16 + n] = c1[r];
    }
    __syncthreads();

    for (int it = tid; it < 64 * 16; it += 256) {
      const int b  = it >> 4;
      const int jj = it & 15;
      const float* xr = xg + ((size_t)(b * T_SZ + t)) * NG + j0 + jj;
      const float gi = gates[b * 64 +  0 + jj] + xr[0 * RNN_D];
      const float gf = gates[b * 64 + 16 + jj] + xr[1 * RNN_D];
      const float gg = gates[b * 64 + 32 + jj] + xr[2 * RNN_D];
      const float go = gates[b * 64 + 48 + jj] + xr[3 * RNN_D];
      const float si = 1.f / (1.f + __expf(-gi));
      const float sf = 1.f / (1.f + __expf(-gf));
      const float so = 1.f / (1.f + __expf(-go));
      const float tg = tanhf(gg);
      const float cn = sf * cst[it] + si * tg;
      const float hn = so * tanhf(cn);
      hs[((size_t)(b * T_SZ + t)) * RNN_D + j0 + jj] = (bf16_t)hn;  // pre-mask
      const float m = 1.f - (float)dones[b * T_SZ + t];
      cst[it] = cn * m;
      hnxt[b * RNN_D + j0 + jj] = (bf16_t)(hn * m);
      // prefetch next timestep's x-gate segments (global_prefetch_b8)
      if (jj == 0 && t + 1 < T_SZ) {
        const float* xn = xr + NG;
#pragma unroll
        for (int g = 0; g < 4; ++g) __builtin_prefetch(xn + g * RNN_D, 0, 1);
      }
    }
    gridsync(sync, NWG_LSTM);   // publish h_next, acquire for next step
  }
}

// ---------------- phase 3: out = hs @ W_out^T + b, fused log_softmax -------
// grid (1023, 13), block 160 (5 waves). nb<12: 64x80 tile (80%5==0 so each
// softmax group of 5 is tile-local); nb==12: done_p column 960 only.
__global__ void out_gemm(const bf16_t* hs, const bf16_t* wout,
                         const float* bout, float* out) {
  __shared__ float tile[64 * 80];
  const int mb = blockIdx.x;
  const int nb = blockIdx.y;
  const int wave = threadIdx.x >> 5;
  const int lane = threadIdx.x & 31;
  const int half = lane >> 4;
  const int n    = lane & 15;
  const int c0 = nb * 80;
  const int nwaves = (nb < 12) ? 5 : 1;

  if (wave < nwaves) {
    const int n0 = c0 + wave * 16;
    v8f acc[4] = {v8f{}, v8f{}, v8f{}, v8f{}};
    for (int k = 0; k < RNN_D; k += 32) {
      v16bf bf = load_bT(wout + (size_t)n0 * RNN_D + k, RNN_D);
#pragma unroll
      for (int mt = 0; mt < 4; ++mt) {
        // A rows remapped: logical m = b*1023 + t  ->  hs row b*1024 + t
        const int ml = mb * 64 + mt * 16 + (lane & 15);
        const int bb = ml / 1023;
        const int tt = ml - bb * 1023;
        const bf16_t* arow = hs + ((size_t)(bb * T_SZ + tt)) * RNN_D + k + half * 16;
        v16bf af;
#pragma unroll
        for (int v = 0; v < 8; ++v) {
          const int kb = 2 * v;
          af[2 * v]     = arow[kb];
          af[2 * v + 1] = arow[kb + 1];
        }
        acc[mt] = wmma_bf16(af, bf, acc[mt]);
      }
    }
    const int col = c0 + wave * 16 + n;
    const float bias = (col < NOUT) ? bout[col] : 0.f;
#pragma unroll
    for (int mt = 0; mt < 4; ++mt)
#pragma unroll
      for (int r = 0; r < 8; ++r)
        tile[(mt * 16 + r + half * 8) * 80 + wave * 16 + n] = acc[mt][r] + bias;
  }
  __syncthreads();

  const size_t SEG = (size_t)MROWS * 320;
  if (nb < 12) {
    for (int it = threadIdx.x; it < 64 * 16; it += 160) {
      const int rr = it >> 4, g = it & 15;
      const size_t ml = (size_t)mb * 64 + rr;
      float v[5];
#pragma unroll
      for (int e = 0; e < 5; ++e) v[e] = tile[rr * 80 + g * 5 + e];
      const int col = c0 + g * 5;
      if (col < 320) {                      // logmix: log_softmax over 5
        float mx = v[0];
#pragma unroll
        for (int e = 1; e < 5; ++e) mx = fmaxf(mx, v[e]);
        float s = 0.f;
#pragma unroll
        for (int e = 0; e < 5; ++e) s += __expf(v[e] - mx);
        const float ls = mx + __logf(s);
        const size_t base = ml * 320 + col;
#pragma unroll
        for (int e = 0; e < 5; ++e) out[base + e] = v[e] - ls;
      } else if (col < 640) {               // mu
        const size_t base = SEG + ml * 320 + (col - 320);
#pragma unroll
        for (int e = 0; e < 5; ++e) out[base + e] = v[e];
      } else {                              // logstd
        const size_t base = 2 * SEG + ml * 320 + (col - 640);
#pragma unroll
        for (int e = 0; e < 5; ++e) out[base + e] = v[e];
      }
    }
  } else {                                  // done_p (column 960)
    for (int rr = threadIdx.x; rr < 64; rr += 160) {
      const size_t ml = (size_t)mb * 64 + rr;
      out[3 * SEG + ml] = tile[rr * 80];
    }
  }
}

// ---------------- host launcher --------------------------------------------

extern "C" void kernel_launch(void* const* d_in, const int* in_sizes, int n_in,
                              void* d_out, int out_size, void* d_ws,
                              size_t ws_size, hipStream_t stream) {
  (void)in_sizes; (void)n_in; (void)out_size; (void)ws_size;
  const float* z       = (const float*)d_in[0];
  const int*   actions = (const int*)d_in[1];
  const int*   dones   = (const int*)d_in[2];
  const float* embed   = (const float*)d_in[3];
  const float* W_ih    = (const float*)d_in[4];
  const float* W_hh    = (const float*)d_in[5];
  const float* b_ih    = (const float*)d_in[6];
  const float* b_hh    = (const float*)d_in[7];
  const float* W_out   = (const float*)d_in[8];
  const float* b_out   = (const float*)d_in[9];
  float* out = (float*)d_out;

  unsigned char* base = (unsigned char*)d_ws;
  size_t off = 0;
  auto take = [&](size_t bytes) -> void* {
    void* p = base + off;
    off += (bytes + 255) & ~(size_t)255;
    return p;
  };
  float*    xg    = (float*)   take((size_t)B_SZ * T_SZ * NG * 4);      // 512MB
  bf16_t*   inp   = (bf16_t*)  take((size_t)B_SZ * T_SZ * KIN * 2);
  bf16_t*   wih   = (bf16_t*)  take((size_t)NG * KIN * 2);
  bf16_t*   whh   = (bf16_t*)  take((size_t)NG * RNN_D * 2);
  bf16_t*   wout  = (bf16_t*)  take((size_t)NOUT_P * RNN_D * 2);
  float*    bcomb = (float*)   take((size_t)NG * 4);
  bf16_t*   hs    = (bf16_t*)  take((size_t)B_SZ * T_SZ * RNN_D * 2);   // 64MB
  bf16_t*   hbuf  = (bf16_t*)  take((size_t)2 * B_SZ * RNN_D * 2);
  unsigned* sync  = (unsigned*)take(256);

  prep_inp<<<B_SZ * T_SZ, KIN, 0, stream>>>(z, actions, embed, inp);
  cvt_pad<<<NG, KIN, 0, stream>>>(W_ih, wih, NG, ZD + 4, KIN);
  cvt_pad<<<NG, RNN_D, 0, stream>>>(W_hh, whh, NG, RNN_D, RNN_D);
  cvt_pad<<<NOUT_P, RNN_D, 0, stream>>>(W_out, wout, NOUT, RNN_D, RNN_D);
  prep_bias<<<(NG + 255) / 256, 256, 0, stream>>>(b_ih, b_hh, bcomb);
  init_state<<<(2 * B_SZ * RNN_D + 255) / 256, 256, 0, stream>>>(hbuf, sync);

  xgates_gemm<<<dim3(1024, 16), 256, 0, stream>>>(inp, wih, bcomb, xg);
  lstm_kernel<<<NWG_LSTM, 256, LSTM_LDS_BYTES, stream>>>(whh, xg, dones, hbuf,
                                                         hs, sync);
  out_gemm<<<dim3(1023, 13), 160, 0, stream>>>(hs, wout, b_out, out);
}
